// Multy_Head_Attention_84954453115548
// MI455X (gfx1250) — matrix operand order repaired and checked by reading, not compiled, not run on hardware
//
#include <hip/hip_runtime.h>

typedef __attribute__((ext_vector_type(16))) __bf16 v16bf;
typedef __attribute__((ext_vector_type(8)))  float  v8f;
typedef int v4i __attribute__((vector_size(16)));
typedef __attribute__((address_space(1))) v4i v4i_g;   // global int4
typedef __attribute__((address_space(3))) v4i v4i_l;   // LDS int4

#define B_  4
#define S_  2048
#define DM  512
#define H_  8
#define KD  64
#define QKV_ELEMS (B_ * H_ * S_ * KD)   // 4,194,304 elems per Q/K/V buffer
#define WT_ELEMS  (24 * KD * DM)        // 786,432
#define HKT_ELEMS (DM * DM)             // 262,144

static __device__ __forceinline__ v8f wmma_bf16(v16bf a, v16bf b, v8f c) {
  return __builtin_amdgcn_wmma_f32_16x16x32_bf16(false, a, false, b, (short)0, c,
                                                 false, false);
}

// A-fragment (16x32, 16-bit) per-lane element -> K index (two 8-wide runs)
static __device__ __forceinline__ int kmapA(int i, int hi) {
  return ((i & 8) << 1) + (hi << 3) + (i & 7);
}

// 16-byte global->LDS copy: CDNA5 async-to-LDS when available, else via VGPR.
static __device__ __forceinline__ void copy16(__bf16* ldsDst, const __bf16* gSrc) {
#if __has_builtin(__builtin_amdgcn_global_load_async_to_lds_b128)
  // global generic addr == AS1 addr; generic LDS addr low 32 bits == LDS offset
  v4i_g* g = (v4i_g*)(uintptr_t)gSrc;
  v4i_l* l = (v4i_l*)(uint32_t)(uintptr_t)ldsDst;
  __builtin_amdgcn_global_load_async_to_lds_b128(g, l, 0, 0);
#else
  *(uint4*)ldsDst = *(const uint4*)gSrc;
#endif
}

static __device__ __forceinline__ void wait_async0() {
#if __has_builtin(__builtin_amdgcn_s_wait_asynccnt)
  __builtin_amdgcn_s_wait_asynccnt(0);
#else
  asm volatile("s_wait_asynccnt 0x0" ::: "memory");
#endif
}

// ---------------------------------------------------------------------------
// Kernel 0: weight prep.  wT[blk][n][k] = kernel[blk][k][n] (bf16),
//           hkT[n][k] = head_kernel[k][n] (bf16).  256 tiles of 64x64.
// ---------------------------------------------------------------------------
__global__ __launch_bounds__(256) void prep_weights_kernel(
    const float* __restrict__ w, const float* __restrict__ hk,
    __bf16* __restrict__ wT, __bf16* __restrict__ hkT) {
  __shared__ float ld[64][65];   // +1 pad: conflict-free transpose
  const int t = blockIdx.x;

  const float* src;
  __bf16* dst;
  int srcStride, dstStride;
  if (t < 192) {                       // qkv weights: blk = t/8, k-tile = t%8
    int blk = t >> 3;
    int kBase = (t & 7) * 64;
    src = w + ((size_t)blk * DM + kBase) * KD;     // 64(k) x 64(n), stride 64
    dst = wT + (size_t)blk * KD * DM + kBase;      // dst[n][k], stride 512
    srcStride = KD;
    dstStride = DM;
  } else {                             // head_kernel 512x512: 8x8 tiles
    int t2 = t - 192;
    int kBase = (t2 & 7) * 64;
    int nBase = (t2 >> 3) * 64;
    src = hk + (size_t)kBase * DM + nBase;
    dst = hkT + (size_t)nBase * DM + kBase;
    srcStride = DM;
    dstStride = DM;
  }

  #pragma unroll
  for (int it = 0; it < 16; ++it) {
    int i = threadIdx.x + it * 256;
    int r = i >> 6, c = i & 63;        // r = k-local, c = n-local
    ld[r][c] = src[(size_t)r * srcStride + c];
  }
  __syncthreads();
  #pragma unroll
  for (int it = 0; it < 16; ++it) {
    int i = threadIdx.x + it * 256;
    int r = i >> 6, c = i & 63;        // r = n-local, c = k-local
    dst[(size_t)r * dstStride + c] = (__bf16)ld[c][r];
  }
}

// ---------------------------------------------------------------------------
// Kernel 1: QKV projection.  Q,K row-major [b][h][s][64]; V transposed
//           [b][h][kf][s].  Grid (24, 128), 128 threads (4 waves).
// ---------------------------------------------------------------------------
__global__ __launch_bounds__(128) void qkv_proj_kernel(
    const float* __restrict__ x, const __bf16* __restrict__ wT,
    __bf16* __restrict__ qk, __bf16* __restrict__ vT) {
  const int tid  = threadIdx.x;
  const int lane = tid & 31;
  const int wv   = tid >> 5;
  const int hi   = lane >> 4;
  const int ln   = lane & 15;

  const int blk = blockIdx.x;            // 0..23 = h*3 + sel
  const int h   = blk / 3;
  const int sel = blk % 3;
  const int rowBase = blockIdx.y * 64;   // row over flattened (B*S)

  __shared__ __align__(16) __bf16 xt[64 * 32];    // [row][k]
  __shared__ __align__(16) __bf16 wtT[64 * 32];   // [n][k]

  v8f acc[4];
  #pragma unroll
  for (int j = 0; j < 4; ++j)
    #pragma unroll
    for (int e = 0; e < 8; ++e) acc[j][e] = 0.0f;

  for (int kk = 0; kk < DM; kk += 32) {
    // stage x tile 64x32 (f32 -> bf16), fixed trip count
    #pragma unroll
    for (int it = 0; it < 4; ++it) {
      int i = tid + it * 128;
      int r = i >> 3;
      int c = (i & 7) * 4;
      const float4 v = *(const float4*)(x + (size_t)(rowBase + r) * DM + kk + c);
      xt[r * 32 + c + 0] = (__bf16)v.x;
      xt[r * 32 + c + 1] = (__bf16)v.y;
      xt[r * 32 + c + 2] = (__bf16)v.z;
      xt[r * 32 + c + 3] = (__bf16)v.w;
    }
    // stage wT tile 64(n) x 32(k): contiguous 16B chunks
    #pragma unroll
    for (int it = 0; it < 2; ++it) {
      int i = tid + it * 128;
      int r = i >> 2;
      int c = (i & 3) * 8;
      copy16(wtT + r * 32 + c, wT + ((size_t)blk * KD + r) * DM + kk + c);
    }
    wait_async0();
    __syncthreads();

    v16bf a;
    #pragma unroll
    for (int i = 0; i < 16; ++i)
      a[i] = xt[(wv * 16 + ln) * 32 + kmapA(i, hi)];

    #pragma unroll
    for (int j = 0; j < 4; ++j) {
      v16bf bf;   // B[k][n] = wT[n][k]: per-lane contiguous in k
      #pragma unroll
      for (int i = 0; i < 16; ++i)
        bf[i] = wtT[(j * 16 + ln) * 32 + kmapA(i, hi)];
      acc[j] = wmma_bf16(a, bf, acc[j]);
    }
    __syncthreads();
  }

  const int b  = rowBase >> 11;
  const int s0 = (rowBase & 2047) + wv * 16 + hi * 8;   // 8 consecutive rows
  if (sel == 2) {
    // V: transposed store [b][h][col][s] — 8 consecutive s -> one uint4
    #pragma unroll
    for (int j = 0; j < 4; ++j) {
      __bf16 p[8];
      #pragma unroll
      for (int r = 0; r < 8; ++r) p[r] = (__bf16)acc[j][r];
      uint4 u;
      __builtin_memcpy(&u, p, 16);
      *(uint4*)(vT + (((size_t)(b * H_ + h)) * KD + j * 16 + ln) * S_ + s0) = u;
    }
  } else {
    const float scl = (sel == 0) ? 0.125f : 1.0f;  // fold 1/sqrt(64) into Q
    __bf16* dst = qk + (size_t)sel * QKV_ELEMS;
    #pragma unroll
    for (int j = 0; j < 4; ++j)
      #pragma unroll
      for (int r = 0; r < 8; ++r)
        dst[(((size_t)(b * H_ + h)) * S_ + s0 + r) * KD + j * 16 + ln] =
            (__bf16)(acc[j][r] * scl);
  }
}

// ---------------------------------------------------------------------------
// Kernel 2: flash attention, double-buffered K/V tiles with async-to-LDS.
// Grid (32 q-tiles, 8 heads, 4 batches), 128 threads (4 waves).
// ---------------------------------------------------------------------------
__global__ __launch_bounds__(128) void attn_kernel(
    const __bf16* __restrict__ qb, const __bf16* __restrict__ kb,
    const __bf16* __restrict__ vbT, __bf16* __restrict__ ob) {
  const int tid  = threadIdx.x;
  const int lane = tid & 31;
  const int wv   = tid >> 5;
  const int hi   = lane >> 4;
  const int ln   = lane & 15;

  const int b  = blockIdx.z;
  const int h  = blockIdx.y;
  const int qt = blockIdx.x * 64;
  const size_t bh   = ((size_t)b * H_ + h) * S_;   // row-major Q/K base rows
  const size_t bh64 = ((size_t)b * H_ + h) * KD;   // transposed V base rows

  __shared__ __align__(16) __bf16 kt[2][64 * KD];   // [t][kf], double buffered
  __shared__ __align__(16) __bf16 vtT[2][KD * 64];  // [kf][t], double buffered
  __shared__ __align__(16) __bf16 pt[4][16 * KD];   // per-wave P pad

  // Q fragments (A layout), per-lane 8-wide contiguous runs
  v16bf qf[2];
  const __bf16* qg = qb + (bh + qt + wv * 16) * KD;
  #pragma unroll
  for (int kk2 = 0; kk2 < 2; ++kk2)
    #pragma unroll
    for (int i = 0; i < 16; ++i)
      qf[kk2][i] = qg[ln * KD + kk2 * 32 + kmapA(i, hi)];

  v8f oacc[4];
  #pragma unroll
  for (int j = 0; j < 4; ++j)
    #pragma unroll
    for (int e = 0; e < 8; ++e) oacc[j][e] = 0.0f;
  float rmax[8], rsum[8];
  #pragma unroll
  for (int r = 0; r < 8; ++r) { rmax[r] = -3.0e38f; rsum[r] = 0.0f; }

  // stage tile tt into buffer buf
  auto stage = [&](int tt, int buf) {
    const int tb = tt * 64;
    #pragma unroll
    for (int it = 0; it < 4; ++it) {           // K tile: contiguous 8KB
      int i = tid + it * 128;
      copy16(&kt[buf][i * 8], kb + (bh + tb) * KD + i * 8);
    }
    #pragma unroll
    for (int it = 0; it < 4; ++it) {           // V tile: 64 rows x 64 t
      int i = tid + it * 128;
      int r = i >> 3;
      int c = (i & 7) * 8;
      copy16(&vtT[buf][r * 64 + c], vbT + (bh64 + r) * S_ + tb + c);
    }
  };

  stage(0, 0);
  wait_async0();
  __syncthreads();

  for (int tt = 0; tt < S_ / 64; ++tt) {
    const int cur = tt & 1;
    if (tt + 1 < S_ / 64) stage(tt + 1, cur ^ 1);   // prefetch next tile

    // S = Q * K^T
    v8f sc[4];
    #pragma unroll
    for (int j = 0; j < 4; ++j) {
      #pragma unroll
      for (int e = 0; e < 8; ++e) sc[j][e] = 0.0f;
      #pragma unroll
      for (int kk2 = 0; kk2 < 2; ++kk2) {
        v16bf bf;   // B[kf][t] = Ktile[t][kf]: per-lane contiguous in kf
        #pragma unroll
        for (int i = 0; i < 16; ++i)
          bf[i] = kt[cur][(j * 16 + ln) * KD + kk2 * 32 + kmapA(i, hi)];
        sc[j] = wmma_bf16(qf[kk2], bf, sc[j]);
      }
    }

    // online softmax
    #pragma unroll
    for (int r = 0; r < 8; ++r) {
      float m0 = fmaxf(fmaxf(sc[0][r], sc[1][r]), fmaxf(sc[2][r], sc[3][r]));
      m0 = fmaxf(m0, __shfl_xor(m0, 1, 32));
      m0 = fmaxf(m0, __shfl_xor(m0, 2, 32));
      m0 = fmaxf(m0, __shfl_xor(m0, 4, 32));
      m0 = fmaxf(m0, __shfl_xor(m0, 8, 32));
      float nm   = fmaxf(rmax[r], m0);
      float corr = __expf(rmax[r] - nm);
      rmax[r] = nm;
      float ps = 0.0f;
      #pragma unroll
      for (int j = 0; j < 4; ++j) {
        float p = __expf(sc[j][r] - nm);
        sc[j][r] = p;
        ps += p;
      }
      ps += __shfl_xor(ps, 1, 32);
      ps += __shfl_xor(ps, 2, 32);
      ps += __shfl_xor(ps, 4, 32);
      ps += __shfl_xor(ps, 8, 32);
      rsum[r] = rsum[r] * corr + ps;
      #pragma unroll
      for (int j = 0; j < 4; ++j) oacc[j][r] *= corr;
      #pragma unroll
      for (int j = 0; j < 4; ++j)
        pt[wv][(r + hi * 8) * KD + j * 16 + ln] = (__bf16)sc[j][r];
    }

    asm volatile("s_wait_dscnt 0x0" ::: "memory");   // per-wave LDS RAW fence

    // P in A-fragment layout (contiguous 8-wide runs)
    v16bf pf[2];
    #pragma unroll
    for (int kk2 = 0; kk2 < 2; ++kk2)
      #pragma unroll
      for (int i = 0; i < 16; ++i)
        pf[kk2][i] = pt[wv][ln * KD + kk2 * 32 + kmapA(i, hi)];

    // O += P * V
    #pragma unroll
    for (int j = 0; j < 4; ++j)
      #pragma unroll
      for (int kk2 = 0; kk2 < 2; ++kk2) {
        v16bf bf;   // B[t][kf] = vtT[kf][t]: per-lane contiguous in t
        #pragma unroll
        for (int i = 0; i < 16; ++i)
          bf[i] = vtT[cur][(j * 16 + ln) * 64 + kk2 * 32 + kmapA(i, hi)];
        oacc[j] = wmma_bf16(pf[kk2], bf, oacc[j]);
      }

    wait_async0();     // next tile landed
    __syncthreads();   // all waves done reading cur, next buffer ready
  }

  // normalize, store concatenated-heads layout (b, s, h*64+kf)
  #pragma unroll
  for (int r = 0; r < 8; ++r) {
    float inv = 1.0f / rsum[r];
    int s = qt + wv * 16 + r + hi * 8;
    #pragma unroll
    for (int j = 0; j < 4; ++j)
      ob[((size_t)(b * S_ + s)) * DM + h * KD + j * 16 + ln] =
          (__bf16)(oacc[j][r] * inv);
  }
}

// ---------------------------------------------------------------------------
// Kernel 3: output projection.  out (8192,512) = o (8192,512) * head_kernel.
// ---------------------------------------------------------------------------
__global__ __launch_bounds__(128) void out_proj_kernel(
    const __bf16* __restrict__ o, const __bf16* __restrict__ hkT,
    float* __restrict__ out) {
  const int tid  = threadIdx.x;
  const int lane = tid & 31;
  const int wv   = tid >> 5;
  const int hi   = lane >> 4;
  const int ln   = lane & 15;

  const int colBase = blockIdx.x * 64;
  const int rowBase = blockIdx.y * 64;

  __shared__ __align__(16) __bf16 at[64 * 32];    // [row][k]
  __shared__ __align__(16) __bf16 wtT[64 * 32];   // [n][k]

  v8f acc[4];
  #pragma unroll
  for (int j = 0; j < 4; ++j)
    #pragma unroll
    for (int e = 0; e < 8; ++e) acc[j][e] = 0.0f;

  for (int kk = 0; kk < DM; kk += 32) {
    #pragma unroll
    for (int it = 0; it < 2; ++it) {
      int i = tid + it * 128;
      int r = i >> 2;
      int c = (i & 3) * 8;
      copy16(at + r * 32 + c, o + (size_t)(rowBase + r) * DM + kk + c);
    }
    #pragma unroll
    for (int it = 0; it < 2; ++it) {
      int i = tid + it * 128;
      int r = i >> 2;
      int c = (i & 3) * 8;
      copy16(wtT + r * 32 + c, hkT + (size_t)(colBase + r) * DM + kk + c);
    }
    wait_async0();
    __syncthreads();

    v16bf a;
    #pragma unroll
    for (int i = 0; i < 16; ++i)
      a[i] = at[(wv * 16 + ln) * 32 + kmapA(i, hi)];

    #pragma unroll
    for (int j = 0; j < 4; ++j) {
      v16bf bf;   // B[k][n] = hkT[n][k]
      #pragma unroll
      for (int i = 0; i < 16; ++i)
        bf[i] = wtT[(j * 16 + ln) * 32 + kmapA(i, hi)];
      acc[j] = wmma_bf16(a, bf, acc[j]);
    }
    __syncthreads();
  }

  #pragma unroll
  for (int j = 0; j < 4; ++j)
    #pragma unroll
    for (int r = 0; r < 8; ++r)
      out[(size_t)(rowBase + wv * 16 + r + hi * 8) * DM + colBase + j * 16 + ln] =
          acc[j][r];
}

// ---------------------------------------------------------------------------
extern "C" void kernel_launch(void* const* d_in, const int* in_sizes, int n_in,
                              void* d_out, int out_size, void* d_ws, size_t ws_size,
                              hipStream_t stream) {
  (void)in_sizes; (void)n_in; (void)out_size; (void)ws_size;
  const float* x    = (const float*)d_in[0];
  const float* kern = (const float*)d_in[1];
  const float* hk   = (const float*)d_in[2];
  float* out        = (float*)d_out;

  __bf16* qbuf  = (__bf16*)d_ws;                      // 8 MB (Q row-major)
  __bf16* kbuf  = qbuf + (size_t)QKV_ELEMS;           // 8 MB (K row-major)
  __bf16* vbufT = kbuf + (size_t)QKV_ELEMS;           // 8 MB (V transposed)
  __bf16* obuf  = vbufT + (size_t)QKV_ELEMS;          // 8 MB (attn out)
  __bf16* wT    = obuf + (size_t)QKV_ELEMS;           // 1.5 MB
  __bf16* hkT   = wT + (size_t)WT_ELEMS;              // 0.5 MB

  prep_weights_kernel<<<256, 256, 0, stream>>>(kern, hk, wT, hkT);
  qkv_proj_kernel<<<dim3(24, 128), 128, 0, stream>>>(x, wT, qbuf, vbufT);
  attn_kernel<<<dim3(32, 8, 4), 128, 0, stream>>>(qbuf, kbuf, vbufT, obuf);
  out_proj_kernel<<<dim3(8, 128), 128, 0, stream>>>(obuf, hkT, out);
}
